// ContextClassifier_43576738185612
// MI455X (gfx1250) — matrix operand order repaired and checked by reading, not compiled, not run on hardware
//
#include <hip/hip_runtime.h>
#include <hip/hip_bf16.h>

typedef __attribute__((ext_vector_type(16))) _Float16 v16h;
typedef __attribute__((ext_vector_type(8)))  _Float16 v8h;
typedef __attribute__((ext_vector_type(8)))  float    v8f;
typedef __attribute__((ext_vector_type(4)))  unsigned int u32x4;
typedef __attribute__((ext_vector_type(8)))  int      i32x8;
typedef __attribute__((ext_vector_type(4)))  int      i32x4;

#define S_LEN 512
#define BATCH 32
#define HID   512
#define NSPAN 2048
#define DDIM  256
#define LDIMC 32
#define LMAXC 16
#define VOC   50257
#define VPAD  50272          // round up to multiple of 32
#define KCTX  1056           // LDIM + 2H  (33 * 32)
#define KPHR  2080           // LDIM + 4H  (65 * 32)

#define NSPLIT 4             // vocab splits (occupancy)
#define MBLK   128           // rows per label block (8 waves x 16 rows)
#define NSTEPS (VPAD / 32)   // 1571 staging steps of 32 vocab columns
#define SPS    ((NSTEPS + NSPLIT - 1) / NSPLIT)   // steps per split = 393

// ---------------- WMMA helpers (CDNA5 16x16x32 f16 -> f32) ----------------

__device__ __forceinline__ v8f wmma_f16(v16h a, v16h b, v8f c) {
    // (neg_a, A, neg_b, B, c_mod, C, reuse_a, reuse_b)
    return __builtin_amdgcn_wmma_f32_16x16x32_f16(false, a, false, b,
                                                  (short)0, c, false, false);
}

// A fragment: 16x32 f16, row-major source with leading dim lda (halves).
// lane<16: M=lane, K = 0..15; lane>=16: M=lane-16, K = 16..31
__device__ __forceinline__ v16h load_a_frag(const _Float16* __restrict__ A,
                                            int lda, int m0, int k, int lane) {
    int row  = m0 + (lane & 15);
    int koff = (lane >> 4) * 16;
    const _Float16* p = A + (size_t)row * lda + k + koff;
    v8h lo = *reinterpret_cast<const v8h*>(p);
    v8h hi = *reinterpret_cast<const v8h*>(p + 8);
    v16h a;
#pragma unroll
    for (int i = 0; i < 8; ++i) { a[i] = lo[i]; a[8 + i] = hi[i]; }
    return a;
}

// B fragment: 32x16 f16 from memory [n][k] row-major (contiguous K per col).
// lane<16: N=lane, K=0..15 ; lane>=16: N=lane-16, K=16..31
__device__ __forceinline__ v16h load_b_frag(const _Float16* __restrict__ W,
                                            int ldb, int n0, int k, int lane) {
    int col  = n0 + (lane & 15);
    int koff = (lane >> 4) * 16;
    const _Float16* p = W + (size_t)col * ldb + k + koff;
    v8h lo = *reinterpret_cast<const v8h*>(p);
    v8h hi = *reinterpret_cast<const v8h*>(p + 8);
    v16h b;
#pragma unroll
    for (int i = 0; i < 8; ++i) { b[i] = lo[i]; b[8 + i] = hi[i]; }
    return b;
}

// Same fragment shape but sourced from LDS (emits ds_load_b128).
__device__ __forceinline__ v16h load_b_frag_lds(const _Float16* sW,
                                                int n0, int k, int lane) {
    int col  = n0 + (lane & 15);
    int koff = (lane >> 4) * 16;
    const _Float16* p = sW + (size_t)col * DDIM + k + koff;
    v8h lo = *reinterpret_cast<const v8h*>(p);
    v8h hi = *reinterpret_cast<const v8h*>(p + 8);
    v16h b;
#pragma unroll
    for (int i = 0; i < 8; ++i) { b[i] = lo[i]; b[8 + i] = hi[i]; }
    return b;
}

// -------- Tensor Data Mover: DMA one 32x256 f16 tile (16 KB) into LDS ------
// D# per CDNA5 ISA ch8: group0 = {count, lds_addr, global_addr, type=2},
// group1 = {data_size=2B, tensor dims == tile dims (32 rows x 256 halves),
// dim0_stride=256}. 2-D tensor -> groups 2/3 zero. Tracked by TENSORcnt.
__device__ __forceinline__ void tdm_load_tile(const _Float16* gsrc,
                                              _Float16* lds_dst) {
    unsigned long long ga = (unsigned long long)(uintptr_t)gsrc;
    unsigned int ldsoff   = (unsigned int)(uintptr_t)lds_dst; // LDS aperture: addr[31:0]
    u32x4 g0;
    g0[0] = 1u;                                        // count=1, user descriptor
    g0[1] = ldsoff;                                    // lds_addr (bytes)
    g0[2] = (unsigned int)(ga & 0xffffffffu);          // global_addr[31:0]
    g0[3] = (unsigned int)((ga >> 32) & 0x01ffffffu)   // global_addr[56:32]
            | (2u << 30);                              // type = 2 ("image")
    i32x8 g1;
    g1[0] = (int)0x00010000;     // workgroup_mask=0, data_size=1 (2 bytes)
    g1[1] = (int)(256u << 16);   // tensor_dim0 = 256 (bits 79:48 lo half)
    g1[2] = (int)(32u  << 16);   // tensor_dim0 hi=0, tensor_dim1 = 32 (lo half)
    g1[3] = (int)(256u << 16);   // tensor_dim1 hi=0, tile_dim0 = 256
    g1[4] = 32;                  // tile_dim1 = 32, tile_dim2 = 0
    g1[5] = 256;                 // tensor_dim0_stride = 256
    g1[6] = 0;                   // dim0_stride hi, dim1_stride lo
    g1[7] = 0;                   // dim1_stride hi
    i32x4 gz = {0, 0, 0, 0};
#if __clang_major__ >= 23
    i32x8 gz8 = {0, 0, 0, 0, 0, 0, 0, 0};
    __builtin_amdgcn_tensor_load_to_lds(g0, g1, gz, gz, gz8, 0);
#else
    __builtin_amdgcn_tensor_load_to_lds(g0, g1, gz, gz, 0);
#endif
}

// ---------------- 1) span gather + f16 pack ----------------

__global__ void ctxcls_pack_kernel(const float* __restrict__ fwd,
                                   const float* __restrict__ bwd,
                                   const int* __restrict__ begins,
                                   const int* __restrict__ ends,
                                   const int* __restrict__ bids,
                                   const float* __restrict__ lemb,
                                   _Float16* __restrict__ ctxA,
                                   _Float16* __restrict__ phrA) {
    int n = blockIdx.x;
    int t = threadIdx.x;
    int beg = begins[n], end = ends[n], bid = bids[n];
    int len = end - beg; if (len > LMAXC) len = LMAXC; len -= 1;
    const float* fb = fwd + ((size_t)(beg - 1) * BATCH + bid) * HID;
    const float* fe = fwd + ((size_t)(end - 1) * BATCH + bid) * HID;
    const float* be = bwd + ((size_t)end * BATCH + bid) * HID;
    const float* bb = bwd + ((size_t)beg * BATCH + bid) * HID;
    _Float16* crow = ctxA + (size_t)n * KCTX;
    _Float16* prow = phrA + (size_t)n * KPHR;
    for (int i = t; i < LDIMC; i += blockDim.x) {
        _Float16 v = (_Float16)lemb[len * LDIMC + i];
        crow[i] = v; prow[i] = v;
    }
    for (int i = t; i < HID; i += blockDim.x) {
        _Float16 vfb = (_Float16)fb[i];
        _Float16 vfe = (_Float16)fe[i];
        _Float16 vbe = (_Float16)be[i];
        _Float16 vbb = (_Float16)bb[i];
        crow[LDIMC + i]            = vfb;   // ctx = [le, f_b, b_e]
        crow[LDIMC + HID + i]      = vbe;
        prow[LDIMC + i]            = vfb;   // phr = [le, f_b, f_e, b_e, b_b]
        prow[LDIMC + HID + i]      = vfe;
        prow[LDIMC + 2 * HID + i]  = vbe;
        prow[LDIMC + 3 * HID + i]  = vbb;
    }
}

// ---------------- 2) fp32 -> f16 weight conversion (with zero padding) -----

__global__ void ctxcls_cvt_kernel(const float* __restrict__ src,
                                  _Float16* __restrict__ dst,
                                  long validN, long totalN) {
    long i = (long)blockIdx.x * blockDim.x + threadIdx.x;
    long stride = (long)gridDim.x * blockDim.x;
    for (; i < totalN; i += stride)
        dst[i] = (i < validN) ? (_Float16)src[i] : (_Float16)0.0f;
}

// ---------------- 3) FFN GEMMs: tanh(A @ W^T + b) -> f16 feats -------------

__global__ __launch_bounds__(256)
void ctxcls_ffn_kernel(const _Float16* __restrict__ ctxA,
                       const _Float16* __restrict__ phrA,
                       const _Float16* __restrict__ wctx,
                       const _Float16* __restrict__ wphr,
                       const float* __restrict__ bctx,
                       const float* __restrict__ bphr,
                       _Float16* __restrict__ feats) {
    int blk  = blockIdx.x;
    int mat  = blk >> 7;           // 0 = ctx, 1 = phr
    int mt   = blk & 127;          // row tile
    int lane = threadIdx.x & 31;
    int wave = threadIdx.x >> 5;
    const _Float16* A  = mat ? phrA : ctxA;
    const _Float16* W  = mat ? wphr : wctx;
    const float* bias  = mat ? bphr : bctx;
    int K      = mat ? KPHR : KCTX;
    int m0     = mt * 16;
    int rowoff = mat * NSPAN;
    for (int ct = wave; ct < DDIM / 16; ct += 8) {
        int n0 = ct * 16;
        v8f acc = {};
        for (int k = 0; k < K; k += 32) {
            v16h a = load_a_frag(A, K, m0, k, lane);
            v16h b = load_b_frag(W, K, n0, k, lane);
            acc = wmma_f16(a, b, acc);
        }
        int   c  = n0 + (lane & 15);
        float bv = bias[c];
        int   rg = (lane >> 4) * 8;
#pragma unroll
        for (int r = 0; r < 8; ++r) {
            float x = tanhf(acc[r] + bv);
            feats[(size_t)(rowoff + m0 + rg + r) * DDIM + c] = (_Float16)x;
        }
    }
}

// ---------------- 4) label GEMM: TDM-staged LDS, fused online log-softmax --
// grid(32, NSPLIT): blockIdx.x -> 128-row group, blockIdx.y -> vocab split.
// 8 waves; wave w owns rows m0+16w (A-fragments resident in 64 VGPRs).
// Wave 0 drives the Tensor Data Mover: one tensor_load_to_lds per 16 KB
// stage of W_lab (32 vocab cols), double-buffered; TENSORcnt + barrier
// hand-off. Every staged byte feeds all 8 waves (8x L2 traffic reduction),
// and the vector pipe stays dedicated to ds_load_b128 -> v_wmma.

__global__ __launch_bounds__(256)
void ctxcls_label_kernel(const _Float16* __restrict__ feats,
                         const _Float16* __restrict__ wlab,
                         const float* __restrict__ blab,
                         const int* __restrict__ tags,
                         float* __restrict__ pmax,
                         float* __restrict__ psum,
                         float* __restrict__ ptgt) {
    __shared__ _Float16 sB[2][32 * DDIM];    // 2 x 16 KB

    int m0    = blockIdx.x * MBLK;
    int split = blockIdx.y;
    int lane  = threadIdx.x & 31;
    int wave  = threadIdx.x >> 5;
    int rg    = (lane >> 4) * 8;
    int m0w   = m0 + wave * 16;              // this wave's 16 rows

    v16h af[8];
#pragma unroll
    for (int kk = 0; kk < 8; ++kk)
        af[kk] = load_a_frag(feats, DDIM, m0w, kk * 32, lane);

    int tg[8];
#pragma unroll
    for (int r = 0; r < 8; ++r)
        tg[r] = tags[(m0w + rg + r) & (NSPAN - 1)];   // tags2 = [tags, tags]

    float rmax[8], rsum[8], tgt[8];
#pragma unroll
    for (int r = 0; r < 8; ++r) {
        rmax[r] = -__builtin_inff(); rsum[r] = 0.0f; tgt[r] = -__builtin_inff();
    }

    int s0 = split * SPS;
    int s1 = s0 + SPS; if (s1 > NSTEPS) s1 = NSTEPS;

    // prologue: DMA step s0 into its buffer, wait, publish
    if (wave == 0) {
        tdm_load_tile(wlab + (size_t)(s0 * 32) * DDIM, sB[s0 & 1]);
        __builtin_amdgcn_s_wait_tensorcnt(0);
    }
    __syncthreads();

    for (int st = s0; st < s1; ++st) {
        int cur  = st & 1;
        bool have = (st + 1) < s1;
        if (wave == 0 && have)   // kick DMA for next stage into idle buffer
            tdm_load_tile(wlab + (size_t)((st + 1) * 32) * DDIM, sB[cur ^ 1]);

        // compute the two 16-column tiles of this stage from LDS
#pragma unroll
        for (int sub = 0; sub < 2; ++sub) {
            v8f acc = {};
#pragma unroll
            for (int kk = 0; kk < 8; ++kk) {
                v16h b = load_b_frag_lds(sB[cur], sub * 16, kk * 32, lane);
                acc = wmma_f16(af[kk], b, acc);
            }
            int c = st * 32 + sub * 16 + (lane & 15);
            if (c < VOC) {
                float bv = blab[c];
#pragma unroll
                for (int r = 0; r < 8; ++r) {
                    float x  = acc[r] + bv;
                    float nm = fmaxf(rmax[r], x);
                    rsum[r]  = rsum[r] * __expf(rmax[r] - nm) + __expf(x - nm);
                    rmax[r]  = nm;
                    if (c == tg[r]) tgt[r] = x;
                }
            }
        }

        if (wave == 0 && have)
            __builtin_amdgcn_s_wait_tensorcnt(0);
        __syncthreads();         // publish next buffer / retire current
    }

    // combine across the 16 columns held by each half-wave (wave32 shuffles)
#pragma unroll
    for (int off = 1; off < 16; off <<= 1) {
#pragma unroll
        for (int r = 0; r < 8; ++r) {
            float om = __shfl_xor(rmax[r], off, 32);
            float os = __shfl_xor(rsum[r], off, 32);
            float ot = __shfl_xor(tgt[r],  off, 32);
            float nm = fmaxf(rmax[r], om);
            rsum[r]  = rsum[r] * __expf(rmax[r] - nm) + os * __expf(om - nm);
            rmax[r]  = nm;
            tgt[r]   = fmaxf(tgt[r], ot);
        }
    }

    // rows are disjoint per wave: lanes 0 / 16 publish this split's partials
    if ((lane & 15) == 0) {
        int base = split * (2 * NSPAN) + m0w + rg;
#pragma unroll
        for (int r = 0; r < 8; ++r) {
            pmax[base + r] = rmax[r];
            psum[base + r] = rsum[r];
            ptgt[base + r] = tgt[r];
        }
    }
}

// ---------------- 5) merge vocab splits -> per-row focal term --------------

__global__ void ctxcls_combine_kernel(const float* __restrict__ pmax,
                                      const float* __restrict__ psum,
                                      const float* __restrict__ ptgt,
                                      float* __restrict__ rowfocal) {
    int row = blockIdx.x * blockDim.x + threadIdx.x;
    if (row >= 2 * NSPAN) return;
    float m = -__builtin_inff(), s = 0.0f, t = -__builtin_inff();
#pragma unroll
    for (int sp = 0; sp < NSPLIT; ++sp) {
        float om = pmax[sp * (2 * NSPAN) + row];
        float os = psum[sp * (2 * NSPAN) + row];
        float nm = fmaxf(m, om);
        s = s * __expf(m - nm) + os * __expf(om - nm);
        m = nm;
        t = fmaxf(t, ptgt[sp * (2 * NSPAN) + row]);
    }
    float lse = m + __logf(s);
    float lp  = t - lse;               // log p_target
    float p   = __expf(lp);
    float q   = 1.0f - p;
    rowfocal[row] = -(q * q) * lp;     // gamma = 2
}

// ---------------- 6) deterministic mean reduction --------------------------

__global__ void ctxcls_reduce_kernel(const float* __restrict__ rowfocal,
                                     float* __restrict__ out) {
    __shared__ float red[256];
    float s = 0.0f;
    for (int i = threadIdx.x; i < 2 * NSPAN; i += 256) s += rowfocal[i];
    red[threadIdx.x] = s;
    __syncthreads();
    for (int off = 128; off > 0; off >>= 1) {
        if ((int)threadIdx.x < off) red[threadIdx.x] += red[threadIdx.x + off];
        __syncthreads();
    }
    if (threadIdx.x == 0) out[0] = red[0] / (2.0f * NSPAN + 1e-5f);
}

// ---------------- host launch ----------------------------------------------

extern "C" void kernel_launch(void* const* d_in, const int* in_sizes, int n_in,
                              void* d_out, int out_size, void* d_ws, size_t ws_size,
                              hipStream_t stream) {
    (void)in_sizes; (void)n_in; (void)out_size; (void)ws_size;
    const float* fwd    = (const float*)d_in[0];
    const float* bwd    = (const float*)d_in[1];
    const int*   begins = (const int*)d_in[2];
    const int*   ends   = (const int*)d_in[3];
    const int*   bids   = (const int*)d_in[4];
    const int*   tags   = (const int*)d_in[5];
    const float* lemb   = (const float*)d_in[6];
    const float* Wctx   = (const float*)d_in[7];
    const float* bctx   = (const float*)d_in[8];
    const float* Wphr   = (const float*)d_in[9];
    const float* bphr   = (const float*)d_in[10];
    const float* Wlab   = (const float*)d_in[11];
    const float* blab   = (const float*)d_in[12];
    float* out = (float*)d_out;

    char* ws = (char*)d_ws;
    size_t off = 0;
    auto carve = [&](size_t bytes) -> void* {
        void* p = (void*)(ws + off);
        off += (bytes + 255) & ~(size_t)255;
        return p;
    };
    _Float16* ctxA   = (_Float16*)carve((size_t)NSPAN * KCTX * 2);
    _Float16* phrA   = (_Float16*)carve((size_t)NSPAN * KPHR * 2);
    _Float16* wctx16 = (_Float16*)carve((size_t)DDIM * KCTX * 2);
    _Float16* wphr16 = (_Float16*)carve((size_t)DDIM * KPHR * 2);
    _Float16* wlab16 = (_Float16*)carve((size_t)VPAD * DDIM * 2);
    _Float16* feats  = (_Float16*)carve((size_t)2 * NSPAN * DDIM * 2);
    float* pmax      = (float*)carve((size_t)NSPLIT * 2 * NSPAN * 4);
    float* psum      = (float*)carve((size_t)NSPLIT * 2 * NSPAN * 4);
    float* ptgt      = (float*)carve((size_t)NSPLIT * 2 * NSPAN * 4);
    float* rowfocal  = (float*)carve((size_t)2 * NSPAN * 4);

    ctxcls_pack_kernel<<<NSPAN, 256, 0, stream>>>(fwd, bwd, begins, ends, bids,
                                                  lemb, ctxA, phrA);
    ctxcls_cvt_kernel<<<256, 256, 0, stream>>>(Wctx, wctx16,
                                               (long)DDIM * KCTX, (long)DDIM * KCTX);
    ctxcls_cvt_kernel<<<256, 256, 0, stream>>>(Wphr, wphr16,
                                               (long)DDIM * KPHR, (long)DDIM * KPHR);
    ctxcls_cvt_kernel<<<1024, 256, 0, stream>>>(Wlab, wlab16,
                                                (long)VOC * DDIM, (long)VPAD * DDIM);
    ctxcls_ffn_kernel<<<256, 256, 0, stream>>>(ctxA, phrA, wctx16, wphr16,
                                               bctx, bphr, feats);
    dim3 lgrid((2 * NSPAN) / MBLK, NSPLIT);
    ctxcls_label_kernel<<<lgrid, 256, 0, stream>>>(feats, wlab16, blab, tags,
                                                   pmax, psum, ptgt);
    ctxcls_combine_kernel<<<(2 * NSPAN) / 256, 256, 0, stream>>>(pmax, psum, ptgt,
                                                                 rowfocal);
    ctxcls_reduce_kernel<<<1, 256, 0, stream>>>(rowfocal, out);
}
